// MDNRNN_80195629351584
// MI455X (gfx1250) — matrix operand order repaired and checked by reading, not compiled, not run on hardware
//
#include <hip/hip_runtime.h>
#include <hip/hip_bf16.h>

typedef __attribute__((ext_vector_type(16))) _Float16 v16h;
typedef __attribute__((ext_vector_type(8)))  _Float16 v8h;
typedef __attribute__((ext_vector_type(8)))  float    v8f;
typedef __attribute__((ext_vector_type(4)))  float    v4f;

#define HID   256
#define TT    512
#define BB    128
#define GATES 1024
#define INF_  35
#define MDN_N 325
#define NPAD  384          // 24 tiles of 16 -> 3 per wave, uniform

// LDS row paddings (bank de-confliction: stride ≡ 4 words mod 64)
#define HPAD  (HID + 8)    // f16 rows: 264 halves = 132 words
#define CPAD  (HID + 4)    // f32 rows: 260 words
#define APAD  20           // act rows: 20 words

// d_out float offsets
#define OFF_LOGPI 0
#define OFF_MU    327680        // 65536*5
#define OFF_SIGMA 10813440      // + 65536*160
#define OFF_C     21299200      // + 65536*160
#define OFF_HF    21331968      // + 128*256

// workspace byte offsets
#define WS_XPROJ  0             // f16 [t][g][b]  512*1024*128
#define WS_HALL   134217728     // f16 [bt][256]  65536*256
#define WS_WHT    167772160     // f16 [n][k]     1024*256
#define WS_WDT    168296448     // f16 [n][k]     384*256 (zero padded)
#define WS_BDP    168493056     // f32 [384]

__device__ __forceinline__ v16h make16(v8h lo, v8h hi) {
  return __builtin_shufflevector(lo, hi, 0,1,2,3,4,5,6,7,8,9,10,11,12,13,14,15);
}

__device__ __forceinline__ float fast_sigmoid(float x) {
  return __builtin_amdgcn_rcpf(1.0f + __expf(-x));
}
__device__ __forceinline__ float fast_tanh(float x) {
  return 2.0f * __builtin_amdgcn_rcpf(1.0f + __expf(-2.0f * x)) - 1.0f;
}

// ---- one-time weight conversion: f32 -> f16, transposed to [n][k] ----
__global__ void convert_kernel(const float* __restrict__ Wh, const float* __restrict__ Wd,
                               const float* __restrict__ bd,
                               _Float16* __restrict__ WhT, _Float16* __restrict__ WdT,
                               float* __restrict__ bdp) {
  int idx = blockIdx.x * 256 + threadIdx.x;
  if (idx < GATES * HID) {            // WhT[n*256+k] = Wh[k][n]
    int n = idx >> 8, k = idx & 255;
    WhT[idx] = (_Float16)Wh[k * GATES + n];
  }
  if (idx < NPAD * HID) {             // WdT[n*256+k] = Wd[k][n], pad n>=325 with 0
    int n = idx >> 8, k = idx & 255;
    float v = (n < MDN_N) ? Wd[k * MDN_N + n] : 0.0f;
    WdT[idx] = (_Float16)v;
  }
  if (idx < NPAD) bdp[idx] = (idx < MDN_N) ? bd[idx] : 0.0f;
}

// ---- xproj[t][g][b] = concat(z,a)[b,t,:] @ Wx[:,g] + bias[g], stored f16 ----
__global__ void xproj_kernel(const float* __restrict__ z, const float* __restrict__ a,
                             const float* __restrict__ Wx, const float* __restrict__ bias,
                             _Float16* __restrict__ xp) {
  __shared__ float xs[BB][INF_ + 1];
  const int t = blockIdx.x;
  const int tid = threadIdx.x;
  for (int i = tid; i < BB * INF_; i += 256) {
    int b = i / INF_, k = i - b * INF_;
    xs[b][k] = (k < 32) ? z[(size_t)(b * TT + t) * 32 + k]
                        : a[(size_t)(b * TT + t) * 3 + (k - 32)];
  }
  __syncthreads();
  const int b  = tid & 127;
  const int g0 = tid >> 7;
  for (int g = g0; g < GATES; g += 2) {
    float acc = bias[g];
#pragma unroll
    for (int k = 0; k < INF_; ++k) acc += xs[b][k] * Wx[k * GATES + g];
    xp[(size_t)(t * GATES + g) * BB + b] = (_Float16)acc;
  }
}

// ---- sequential LSTM: 8 WGs (16 batch rows each), 512 threads = 16 waves ----
__global__ void __launch_bounds__(512, 1)
lstm_kernel(const _Float16* __restrict__ xp, const _Float16* __restrict__ WhT,
            _Float16* __restrict__ hall, float* __restrict__ out) {
  __shared__ _Float16 hs[16][HPAD];    // h state (WMMA A source), padded
  __shared__ float    cs[16][CPAD];    // c state, padded
  __shared__ float    act[GATES][APAD];// activated gates [col][m], padded
  const int tid = threadIdx.x;
  const int m0  = blockIdx.x * 16;

  for (int i = tid; i < 16 * HPAD; i += 512) ((_Float16*)hs)[i] = (_Float16)0.0f;
  for (int i = tid; i < 16 * CPAD; i += 512) ((float*)cs)[i] = 0.0f;

  const int wave = tid >> 5;
  const int lane = tid & 31;
  const int lh   = lane >> 4;   // K-half selector
  const int ll   = lane & 15;   // M (for A) / N (for B,C)
  const int ntg0 = wave * 4;    // first of this wave's 4 N-tiles

  const _Float16* bbase = WhT + (size_t)(ntg0 * 16 + ll) * HID + lh * 8;

  // B fragments for ks=0..3 resident in registers across all 512 steps (128 VGPRs)
  v16h Bres[16];
#pragma unroll
  for (int ks = 0; ks < 4; ++ks)
#pragma unroll
    for (int nt = 0; nt < 4; ++nt) {
      const _Float16* bq = bbase + ks * 32 + (size_t)nt * 16 * HID;
      Bres[ks * 4 + nt] = make16(*(const v8h*)bq, *(const v8h*)(bq + 16));
    }

  for (int t = 0; t < TT; ++t) {
    __syncthreads();  // h/c state for this step is ready

    // prefetch this step's input-projection values (L2) under the WMMA chain
    v8h xv[4];
#pragma unroll
    for (int nt = 0; nt < 4; ++nt)
      xv[nt] = *(const v8h*)(xp + ((size_t)t * GATES + (ntg0 + nt) * 16 + ll) * BB +
                             m0 + lh * 8);

    // opaque *offset* (keeps addrspace-1 provenance -> global_load, not flat_load;
    // still defeats loop-invariant hoisting of the streamed B fragments)
    unsigned soff = 4 * 32;
    asm volatile("" : "+v"(soff));
    const _Float16* bs = bbase + soff;

    v8f acc[4] = {};
    // K-steps 0..3: A from LDS, B register-resident
#pragma unroll
    for (int ks = 0; ks < 4; ++ks) {
      const _Float16* ap = &hs[ll][ks * 32 + lh * 8];
      v16h A = make16(*(const v8h*)ap, *(const v8h*)(ap + 16));
#pragma unroll
      for (int nt = 0; nt < 4; ++nt)
        acc[nt] = __builtin_amdgcn_wmma_f32_16x16x32_f16(false, A, false, Bres[ks * 4 + nt],
                                                         (short)0, acc[nt], false, false);
    }
    // K-steps 4..7: A from LDS, B streamed from L2 (LOADcnt-only pipeline)
#pragma unroll
    for (int ks = 0; ks < 4; ++ks) {
      const _Float16* ap = &hs[ll][(ks + 4) * 32 + lh * 8];
      v16h A = make16(*(const v8h*)ap, *(const v8h*)(ap + 16));
      const _Float16* bp = bs + ks * 32;
#pragma unroll
      for (int nt = 0; nt < 4; ++nt) {
        const _Float16* bq = bp + (size_t)nt * 16 * HID;
        v16h B = make16(*(const v8h*)bq, *(const v8h*)(bq + 16));
        acc[nt] = __builtin_amdgcn_wmma_f32_16x16x32_f16(false, A, false, B,
                                                         (short)0, acc[nt], false, false);
      }
    }

#pragma unroll
    for (int nt = 0; nt < 4; ++nt) {
      const int ntg = ntg0 + nt;
      const int gate = ntg >> 4;              // 0=i 1=f 2=g 3=o
      float res[8];
#pragma unroll
      for (int r = 0; r < 8; ++r) {
        float v = acc[nt][r] + (float)xv[nt][r];
        res[r] = (gate == 2) ? fast_tanh(v) : fast_sigmoid(v);
      }
      float* adst = &act[ntg * 16 + ll][lh * 8];
      *(v4f*)adst       = (v4f){res[0], res[1], res[2], res[3]};
      *(v4f*)(adst + 4) = (v4f){res[4], res[5], res[6], res[7]};
    }
    __syncthreads();  // gates ready

    // elementwise state update: m = tid&15 keeps act/cs accesses conflict-free
    {
      const int m  = tid & 15;
      const int hb = (tid >> 4) * 8;
      _Float16 htmp[8];
#pragma unroll
      for (int j = 0; j < 8; ++j) {
        const int hx = hb + j;
        float iv = act[hx][m];
        float fv = act[256 + hx][m];
        float gv = act[512 + hx][m];
        float ov = act[768 + hx][m];
        float c  = fv * cs[m][hx] + iv * gv;
        float h  = ov * fast_tanh(c);
        cs[m][hx] = c;
        hs[m][hx] = (_Float16)h;
        htmp[j]   = (_Float16)h;
      }
      *(v8h*)&hall[((size_t)(m0 + m) * TT + t) * HID + hb] = *(v8h*)htmp;
    }
  }

  // final carry (c, h) from LDS state
  __syncthreads();
  for (int i = tid; i < 16 * HID; i += 512) {
    int m = i >> 8, hx = i & 255;
    out[OFF_C  + (m0 + m) * HID + hx] = cs[m][hx];
    out[OFF_HF + (m0 + m) * HID + hx] = (float)hs[m][hx];
  }
}

// ---- MDN head: out(65536,325) = hall(65536,256) @ Wd + bd, fused epilogue ----
__global__ void __launch_bounds__(256)
mdn_kernel(const _Float16* __restrict__ hall, const _Float16* __restrict__ WdT,
           const float* __restrict__ bdp, float* __restrict__ out) {
  __shared__ _Float16 hsh[16][HPAD];  // A tile, padded
  __shared__ float    pish[16][8];
  const int tid = threadIdx.x;
  const size_t row0 = (size_t)blockIdx.x * 16;
  {
    const int r  = tid >> 4;
    const int cb = (tid & 15) * 16;
    *(v8h*)&hsh[r][cb]     = *(const v8h*)&hall[(row0 + r) * HID + cb];
    *(v8h*)&hsh[r][cb + 8] = *(const v8h*)&hall[(row0 + r) * HID + cb + 8];
  }
  __syncthreads();

  const int wave = tid >> 5, lane = tid & 31;
  const int lh = lane >> 4, ll = lane & 15;
  const int ntg0 = wave * 3;          // 8 waves x 3 tiles = 24 tiles = 384 cols

  v8f acc[3] = {};
  const _Float16* bbase = WdT + (size_t)(ntg0 * 16 + ll) * HID + lh * 8;
#pragma unroll
  for (int ks = 0; ks < 8; ++ks) {
    const _Float16* ap = &hsh[ll][ks * 32 + lh * 8];
    v16h A = make16(*(const v8h*)ap, *(const v8h*)(ap + 16));
    const _Float16* bp = bbase + ks * 32;
#pragma unroll
    for (int j = 0; j < 3; ++j) {
      const _Float16* bq = bp + (size_t)j * 16 * HID;
      v16h B = make16(*(const v8h*)bq, *(const v8h*)(bq + 16));
      acc[j] = __builtin_amdgcn_wmma_f32_16x16x32_f16(false, A, false, B,
                                                      (short)0, acc[j], false, false);
    }
  }

#pragma unroll
  for (int j = 0; j < 3; ++j) {
    const int n = (ntg0 + j) * 16 + ll;
    const float bias = bdp[n];
#pragma unroll
    for (int r = 0; r < 8; ++r) {
      const int m = r + lh * 8;
      const size_t bt = row0 + m;
      float v = acc[j][r] + bias;
      if (n < 5)                 pish[m][n] = v;                                  // pi logits
      else if (n < 5 + 160)      out[OFF_MU + bt * 160 + (n - 5)] = v;            // mu
      else if (n < MDN_N)        out[OFF_SIGMA + bt * 160 + (n - 165)] = __expf(v) + 1e-6f;
      // n >= 325: padding, drop
    }
  }
  __syncthreads();
  if (tid < 16) {   // per-row log-softmax over the 5 pi logits
    float mx = -1e30f;
#pragma unroll
    for (int j = 0; j < 5; ++j) mx = fmaxf(mx, pish[tid][j]);
    float s = 0.0f;
#pragma unroll
    for (int j = 0; j < 5; ++j) s += __expf(pish[tid][j] - mx);
    const float lse = mx + __logf(s);
    const size_t bt = row0 + tid;
#pragma unroll
    for (int j = 0; j < 5; ++j) out[OFF_LOGPI + bt * 5 + j] = pish[tid][j] - lse;
  }
}

extern "C" void kernel_launch(void* const* d_in, const int* in_sizes, int n_in,
                              void* d_out, int out_size, void* d_ws, size_t ws_size,
                              hipStream_t stream) {
  const float* z  = (const float*)d_in[0];
  const float* a  = (const float*)d_in[1];
  const float* Wx = (const float*)d_in[2];
  const float* Wh = (const float*)d_in[3];
  const float* b  = (const float*)d_in[4];
  const float* Wd = (const float*)d_in[5];
  const float* bd = (const float*)d_in[6];
  float* out = (float*)d_out;
  char* wsb = (char*)d_ws;
  _Float16* xp   = (_Float16*)(wsb + WS_XPROJ);
  _Float16* hall = (_Float16*)(wsb + WS_HALL);
  _Float16* WhT  = (_Float16*)(wsb + WS_WHT);
  _Float16* WdT  = (_Float16*)(wsb + WS_WDT);
  float*    bdp  = (float*)   (wsb + WS_BDP);

  convert_kernel<<<(GATES * HID) / 256, 256, 0, stream>>>(Wh, Wd, bd, WhT, WdT, bdp);
  xproj_kernel<<<TT, 256, 0, stream>>>(z, a, Wx, b, xp);
  lstm_kernel<<<BB / 16, 512, 0, stream>>>(xp, WhT, hall, out);
  mdn_kernel<<<(BB * TT) / 16, 256, 0, stream>>>(hall, WdT, bdp, out);
}
